// GTRANBRel_1666447311030
// MI455X (gfx1250) — matrix operand order repaired and checked by reading, not compiled, not run on hardware
//
#include <hip/hip_runtime.h>
#include <math.h>

// ---------------------------------------------------------------------------
// Types for CDNA5 WMMA (wave32): v_wmma_f32_16x16x32_f16
// ---------------------------------------------------------------------------
typedef __attribute__((ext_vector_type(16))) _Float16 v16h;
typedef __attribute__((ext_vector_type(8)))  _Float16 v8h;
typedef __attribute__((ext_vector_type(8)))  float    v8f;

#define HD 256   // H*D width
#define NH 8     // heads
#define DD 32    // head dim
#define RELMAX 64
#define SCALE_ATT 0.17677669529663687f  // 1/sqrt(32)

// ---------------------------------------------------------------------------
// float ordered-key helpers for atomicMax-based segment max
// ---------------------------------------------------------------------------
__device__ __forceinline__ unsigned fkey(float f) {
  unsigned u = __float_as_uint(f);
  return (u & 0x80000000u) ? ~u : (u | 0x80000000u);
}
__device__ __forceinline__ float funkey(unsigned u) {
  u = (u & 0x80000000u) ? (u & 0x7fffffffu) : ~u;
  return __uint_as_float(u);
}

// ---------------------------------------------------------------------------
// f32 -> f16 conversion (row-major, for the A operand)
// ---------------------------------------------------------------------------
__global__ void cvt_f32_f16_kernel(const float* __restrict__ src,
                                   _Float16* __restrict__ dst, long n) {
  long i = (long)blockIdx.x * blockDim.x + threadIdx.x;
  if (i < n) dst[i] = (_Float16)src[i];
}

// f32 [K x N] -> f16 transposed [N x K]  (B operand: column contiguous in K)
__global__ void cvt_transpose_f16_kernel(const float* __restrict__ W,
                                         _Float16* __restrict__ Wt,
                                         int K, int Ncol) {
  long i = (long)blockIdx.x * blockDim.x + threadIdx.x;
  if (i >= (long)K * Ncol) return;
  int n  = (int)(i / K);
  int kk = (int)(i - (long)n * K);
  Wt[(size_t)n * K + kk] = (_Float16)W[(size_t)kk * Ncol + n];
}

__global__ void zero_u32_kernel(unsigned* __restrict__ p, long n) {
  long i = (long)blockIdx.x * blockDim.x + threadIdx.x;
  if (i < n) p[i] = 0u;
}

// ---------------------------------------------------------------------------
// relp[r][c] = sum_k rel_emb[r][k] * We[k][c]   (64 x 32 @ 32 x 256, f32)
// ---------------------------------------------------------------------------
__global__ void relproj_kernel(const float* __restrict__ rel_emb,
                               const float* __restrict__ We,
                               float* __restrict__ relp, int reln) {
  int tid = blockIdx.x * blockDim.x + threadIdx.x;
  if (tid >= reln * HD) return;
  int r = tid >> 8, c = tid & 255;
  float s = 0.f;
#pragma unroll
  for (int kk = 0; kk < 32; ++kk) s += rel_emb[r * 32 + kk] * We[kk * HD + c];
  relp[tid] = s;
}

// ---------------------------------------------------------------------------
// WMMA GEMM: C[M x Ncol] = A[M x K] @ B[K x Ncol], f16 in / f32 out.
// B is supplied pre-transposed: Bt[Ncol x K].
// One wave32 computes a 16x64 strip (4 tiles): the A fragment is loaded once
// (two b128 loads) and reused across 4 WMMAs per K-step; each B fragment is
// two contiguous b128 loads thanks to the transpose.
// Fragment K order (A 16x32, B 32x16): contiguous K = half*16 + [0..15], same in both; f32 C per ISA.
// ---------------------------------------------------------------------------
__global__ void wmma_gemm_f16_kernel(const _Float16* __restrict__ A,
                                     const _Float16* __restrict__ Bt,
                                     float* __restrict__ C,
                                     int M, int K, int Ncol) {
  const int gwave   = (int)(((long)blockIdx.x * blockDim.x + threadIdx.x) >> 5);
  const int lane    = threadIdx.x & 31;
  const int stripsN = Ncol >> 6;               // 64-wide output strips
  const int tm      = gwave / stripsN;
  const int ts      = gwave - tm * stripsN;
  if (tm * 16 >= M) return;                    // wave-uniform: EXEC stays all-1s

  const int half = lane >> 4;                  // 0: lanes 0-15, 1: lanes 16-31
  const int l16  = lane & 15;
  const _Float16* Ar = A + (size_t)(tm * 16 + l16) * K;

  v8f acc[4] = {{}, {}, {}, {}};
  for (int k0 = 0; k0 < K; k0 += 32) {
    if (k0 + 32 < K)   // speculative prefetch of next A chunk (global_prefetch_b8)
      __builtin_prefetch(Ar + k0 + 32, 0, 1);

    // A 16x32 f16 fragment: K = half*16 + [0..15] contiguous
    v8h a0 = *(const v8h*)(Ar + k0 + half * 16);
    v8h a1 = *(const v8h*)(Ar + k0 + half * 16 + 8);
    v16h a = __builtin_shufflevector(a0, a1, 0, 1, 2, 3, 4, 5, 6, 7,
                                     8, 9, 10, 11, 12, 13, 14, 15);
#pragma unroll
    for (int t = 0; t < 4; ++t) {
      // B 32x16 f16 fragment: lane column = strip + t*16 + l16,
      // K = half*16 + [0..15] contiguous in Bt
      const _Float16* Bc =
          Bt + (size_t)(ts * 64 + t * 16 + l16) * K + k0 + half * 16;
      v8h b0 = *(const v8h*)(Bc);
      v8h b1 = *(const v8h*)(Bc + 8);
      v16h b = __builtin_shufflevector(b0, b1, 0, 1, 2, 3, 4, 5, 6, 7,
                                       8, 9, 10, 11, 12, 13, 14, 15);
      acc[t] = __builtin_amdgcn_wmma_f32_16x16x32_f16(
          /*neg_a=*/false, a, /*neg_b=*/false, b,
          /*c_mod=*/(short)0, acc[t], /*reuse_a=*/false, /*reuse_b=*/false);
    }
  }
  // C 16x16 f32 tiles: VGPR r -> M = r + half*8, N = lane&15
#pragma unroll
  for (int t = 0; t < 4; ++t) {
#pragma unroll
    for (int r = 0; r < 8; ++r) {
      int m = tm * 16 + r + half * 8;
      C[(size_t)m * Ncol + ts * 64 + t * 16 + l16] = acc[t][r];
    }
  }
}

// ---------------------------------------------------------------------------
// Edge pass 1: scores[e][h] = SCALE * <q[dst], k[src]+relp[et]>, segment max.
// Grid: (edges, head). The 64x32 relp slice for this head is staged in LDS.
// ---------------------------------------------------------------------------
__global__ void edge_score_kernel(const float* __restrict__ q,
                                  const float* __restrict__ k,
                                  const float* __restrict__ relp,
                                  const int* __restrict__ src,
                                  const int* __restrict__ dst,
                                  const int* __restrict__ et,
                                  float* __restrict__ scores,
                                  unsigned* __restrict__ mkey,
                                  int E_, int reln) {
  const int h = blockIdx.y;
  __shared__ float s_relp[RELMAX * DD];        // 8 KB
  for (int i = threadIdx.x; i < reln * DD; i += blockDim.x)
    s_relp[i] = relp[(size_t)(i >> 5) * HD + h * DD + (i & 31)];
  __syncthreads();

  int e = blockIdx.x * blockDim.x + threadIdx.x;
  if (e >= E_) return;
  int d = dst[e];
  const float4* qp = (const float4*)(q + (size_t)d * HD + h * DD);
  const float4* kp = (const float4*)(k + (size_t)src[e] * HD + h * DD);
  const float4* ep = (const float4*)(s_relp + et[e] * DD);
  float s = 0.f;
#pragma unroll
  for (int i = 0; i < 8; ++i) {
    float4 qv = qp[i], kv = kp[i], ev = ep[i];
    s += qv.x * (kv.x + ev.x) + qv.y * (kv.y + ev.y) +
         qv.z * (kv.z + ev.z) + qv.w * (kv.w + ev.w);
  }
  s *= SCALE_ATT;
  scores[(size_t)e * NH + h] = s;
  atomicMax(&mkey[d * NH + h], fkey(s));
}

// ---------------------------------------------------------------------------
// Per (node,head): decode max, zero denom, zero agg slice
// ---------------------------------------------------------------------------
__global__ void seg_finalize_kernel(const unsigned* __restrict__ mkey,
                                    float* __restrict__ m,
                                    float* __restrict__ denom,
                                    float* __restrict__ agg, int Nn) {
  int tid = blockIdx.x * blockDim.x + threadIdx.x;
  if (tid >= Nn * NH) return;
  unsigned key = mkey[tid];
  float mv = (key == 0u) ? 0.f : funkey(key);
  if (!__builtin_isfinite(mv)) mv = 0.f;       // guard empty segments
  m[tid] = mv;
  denom[tid] = 0.f;
  int n = tid >> 3, h = tid & 7;
  float4* a = (float4*)(agg + (size_t)n * HD + h * DD);
#pragma unroll
  for (int i = 0; i < 8; ++i) a[i] = make_float4(0.f, 0.f, 0.f, 0.f);
}

// ---------------------------------------------------------------------------
// Edge pass 2: ex = exp(score - m[dst]); denom += ex
// ---------------------------------------------------------------------------
__global__ void edge_exp_kernel(const int* __restrict__ dst,
                                const float* __restrict__ m,
                                float* __restrict__ scores,
                                float* __restrict__ denom, int E_) {
  int tid = blockIdx.x * blockDim.x + threadIdx.x;
  if (tid >= E_ * NH) return;
  int e = tid >> 3, h = tid & 7;
  float ex = __expf(scores[tid] - m[dst[e] * NH + h]);
  scores[tid] = ex;
  __hip_atomic_fetch_add(&denom[dst[e] * NH + h], ex,
                         __ATOMIC_RELAXED, __HIP_MEMORY_SCOPE_AGENT);
}

// ---------------------------------------------------------------------------
// Edge pass 3: agg[dst] += alpha * (v[src] + relp[et]).  Grid: (edges, head),
// relp slice staged in LDS.
// ---------------------------------------------------------------------------
__global__ void edge_scatter_kernel(const float* __restrict__ v,
                                    const float* __restrict__ relp,
                                    const int* __restrict__ src,
                                    const int* __restrict__ dst,
                                    const int* __restrict__ et,
                                    const float* __restrict__ scores,
                                    const float* __restrict__ denom,
                                    float* __restrict__ agg,
                                    int E_, int reln) {
  const int h = blockIdx.y;
  __shared__ float s_relp[RELMAX * DD];        // 8 KB
  for (int i = threadIdx.x; i < reln * DD; i += blockDim.x)
    s_relp[i] = relp[(size_t)(i >> 5) * HD + h * DD + (i & 31)];
  __syncthreads();

  int e = blockIdx.x * blockDim.x + threadIdx.x;
  if (e >= E_) return;
  int d = dst[e];
  float alpha = scores[(size_t)e * NH + h] / (denom[d * NH + h] + 1e-9f);
  const float4* vp = (const float4*)(v + (size_t)src[e] * HD + h * DD);
  const float4* ep = (const float4*)(s_relp + et[e] * DD);
  float* a = agg + (size_t)d * HD + h * DD;
#pragma unroll
  for (int i = 0; i < 8; ++i) {
    float4 vv = vp[i], ev = ep[i];
    __hip_atomic_fetch_add(&a[4 * i + 0], alpha * (vv.x + ev.x),
                           __ATOMIC_RELAXED, __HIP_MEMORY_SCOPE_AGENT);
    __hip_atomic_fetch_add(&a[4 * i + 1], alpha * (vv.y + ev.y),
                           __ATOMIC_RELAXED, __HIP_MEMORY_SCOPE_AGENT);
    __hip_atomic_fetch_add(&a[4 * i + 2], alpha * (vv.z + ev.z),
                           __ATOMIC_RELAXED, __HIP_MEMORY_SCOPE_AGENT);
    __hip_atomic_fetch_add(&a[4 * i + 3], alpha * (vv.w + ev.w),
                           __ATOMIC_RELAXED, __HIP_MEMORY_SCOPE_AGENT);
  }
}

// ---------------------------------------------------------------------------
// Node output: h = elu(agg + res); also emit f16 copy for next layer's GEMM
// ---------------------------------------------------------------------------
__global__ void node_out_kernel(const float* __restrict__ agg,
                                const float* __restrict__ res,
                                float* __restrict__ h,
                                _Float16* __restrict__ hb, long n) {
  long i = (long)blockIdx.x * blockDim.x + threadIdx.x;
  if (i >= n) return;
  float x_ = agg[i] + res[i];
  float y = x_ > 0.f ? x_ : (__expf(x_) - 1.f);
  h[i] = y;
  hb[i] = (_Float16)y;
}

// ---------------------------------------------------------------------------
// Final: out[n] = relu((centrality*gamma+beta) * (h[n] . Wo + bo))
// ---------------------------------------------------------------------------
__global__ void final_kernel(const float* __restrict__ h,
                             const float* __restrict__ Wo,
                             const float* __restrict__ bo,
                             const float* __restrict__ cen,
                             const float* __restrict__ gamma,
                             const float* __restrict__ beta,
                             float* __restrict__ out, int Nn) {
  int n = blockIdx.x * blockDim.x + threadIdx.x;
  if (n >= Nn) return;
  const float4* hp = (const float4*)(h + (size_t)n * HD);
  const float4* wp = (const float4*)Wo;
  float s = 0.f;
#pragma unroll
  for (int i = 0; i < HD / 4; ++i) {
    float4 hv = hp[i], wv = wp[i];
    s += hv.x * wv.x + hv.y * wv.y + hv.z * wv.z + hv.w * wv.w;
  }
  float logit = s + bo[0];
  float sc = cen[n] * gamma[0] + beta[0];
  out[n] = fmaxf(sc * logit, 0.f);
}

// ---------------------------------------------------------------------------
// Host launcher
// ---------------------------------------------------------------------------
extern "C" void kernel_launch(void* const* d_in, const int* in_sizes, int n_in,
                              void* d_out, int out_size, void* d_ws, size_t ws_size,
                              hipStream_t stream) {
  const float* x     = (const float*)d_in[0];
  const float* cen   = (const float*)d_in[1];
  const float* rel   = (const float*)d_in[2];
  const float* Wf[10] = {
      (const float*)d_in[3],  (const float*)d_in[4],  (const float*)d_in[5],
      (const float*)d_in[7],                              // Wq1 Wk1 Wv1 Wr1
      (const float*)d_in[8],  (const float*)d_in[9],  (const float*)d_in[10],
      (const float*)d_in[12],                             // Wq2 Wk2 Wv2 Wr2
      (const float*)d_in[6],  (const float*)d_in[11] };   // We1 We2
  const float* Wo    = (const float*)d_in[13];
  const float* bo    = (const float*)d_in[14];
  const float* gamma = (const float*)d_in[15];
  const float* beta  = (const float*)d_in[16];
  const int* esrc    = (const int*)d_in[17];
  const int* edst    = (const int*)d_in[18];
  const int* etyp    = (const int*)d_in[19];
  float* out = (float*)d_out;

  const int Nn   = in_sizes[1];          // 20000
  const int Ee   = in_sizes[17];         // 200000
  const int INd  = in_sizes[0] / Nn;     // 128
  const int RELN = in_sizes[2] / 32;     // 64

  // ---- workspace carve-out ----
  char* p = (char*)d_ws;
  auto take = [&](size_t bytes) -> char* {
    char* r = p; p += (bytes + 255) & ~(size_t)255; return r;
  };
  _Float16* xb = (_Float16*)take((size_t)Nn * INd * 2);
  _Float16* hb = (_Float16*)take((size_t)Nn * HD * 2);
  _Float16* wb[8];   // transposed f16 weights: [HD x K]
  for (int i = 0; i < 4; ++i) wb[i]     = (_Float16*)take((size_t)HD * INd * 2);
  for (int i = 0; i < 4; ++i) wb[4 + i] = (_Float16*)take((size_t)HD * HD * 2);
  float* q    = (float*)take((size_t)Nn * HD * 4);
  float* k    = (float*)take((size_t)Nn * HD * 4);
  float* v    = (float*)take((size_t)Nn * HD * 4);
  float* rr   = (float*)take((size_t)Nn * HD * 4);
  float* agg  = (float*)take((size_t)Nn * HD * 4);
  float* hf   = (float*)take((size_t)Nn * HD * 4);
  float* relp = (float*)take((size_t)RELN * HD * 4);
  float* sc   = (float*)take((size_t)Ee * NH * 4);
  unsigned* mkey = (unsigned*)take((size_t)Nn * NH * 4);
  float* mval = (float*)take((size_t)Nn * NH * 4);
  float* den  = (float*)take((size_t)Nn * NH * 4);

  const int T = 256;
  auto nb = [](long n, int t) { return (int)((n + t - 1) / t); };

  // ---- convert inputs (row-major) / weights (transposed) to f16 ----
  cvt_f32_f16_kernel<<<nb((long)Nn * INd, T), T, 0, stream>>>(x, xb, (long)Nn * INd);
  for (int i = 0; i < 4; ++i)
    cvt_transpose_f16_kernel<<<nb((long)INd * HD, T), T, 0, stream>>>(Wf[i], wb[i], INd, HD);
  for (int i = 0; i < 4; ++i)
    cvt_transpose_f16_kernel<<<nb((long)HD * HD, T), T, 0, stream>>>(Wf[4 + i], wb[4 + i], HD, HD);

  auto gemm = [&](const _Float16* A, const _Float16* Bt, float* C, int M, int K) {
    long waves = (long)((M + 15) >> 4) * (HD >> 6);   // 16x64 strips
    wmma_gemm_f16_kernel<<<nb(waves * 32, T), T, 0, stream>>>(A, Bt, C, M, K, HD);
  };

  auto layer = [&](const _Float16* hin, int K, int wbase, const float* We) {
    // relation projection table (64 x 256) replaces the [E,256] e tensor
    relproj_kernel<<<nb((long)RELN * HD, T), T, 0, stream>>>(rel, We, relp, RELN);
    gemm(hin, wb[wbase + 0], q,  Nn, K);
    gemm(hin, wb[wbase + 1], k,  Nn, K);
    gemm(hin, wb[wbase + 2], v,  Nn, K);
    gemm(hin, wb[wbase + 3], rr, Nn, K);
    zero_u32_kernel<<<nb((long)Nn * NH, T), T, 0, stream>>>(mkey, (long)Nn * NH);
    dim3 eg(nb(Ee, T), NH);
    edge_score_kernel<<<eg, T, 0, stream>>>(q, k, relp, esrc, edst, etyp, sc, mkey, Ee, RELN);
    seg_finalize_kernel<<<nb((long)Nn * NH, T), T, 0, stream>>>(mkey, mval, den, agg, Nn);
    edge_exp_kernel<<<nb((long)Ee * NH, T), T, 0, stream>>>(edst, mval, sc, den, Ee);
    edge_scatter_kernel<<<eg, T, 0, stream>>>(v, relp, esrc, edst, etyp, sc, den, agg, Ee, RELN);
    node_out_kernel<<<nb((long)Nn * HD, T), T, 0, stream>>>(agg, rr, hf, hb, (long)Nn * HD);
  };

  layer(xb, INd, 0, Wf[8]);   // layer 1: x[128] -> h[256]
  layer(hb, HD,  4, Wf[9]);   // layer 2: h[256] -> h[256]

  final_kernel<<<nb(Nn, T), T, 0, stream>>>(hf, Wo, bo, cen, gamma, beta, out, Nn);
}
